// SimpleRNN_37426345017535
// MI455X (gfx1250) — matrix operand order repaired and checked by reading, not compiled, not run on hardware
//
#include <hip/hip_runtime.h>
#include <hip/hip_fp16.h>

#define Bsz  256
#define Tn   2048
#define DIN  10
#define UN   64
#define DOUT 2
#define HP   72   // LDS row pitch in halves (64 + 8 pad -> conflict-free banks)

typedef __attribute__((ext_vector_type(16))) _Float16 v16h;
typedef __attribute__((ext_vector_type(8)))  float    v8f;
typedef __attribute__((ext_vector_type(2)))  float    v2f;

union AHalf16 { uint4 q[2]; v16h v; };
union H8      { uint4 q;    _Float16 h[8]; };

__device__ __forceinline__ float fast_tanh(float v) {
#if __has_builtin(__builtin_amdgcn_tanhf)
  return __builtin_amdgcn_tanhf(v);
#elif __has_builtin(__builtin_amdgcn_tanh_f32)
  return __builtin_amdgcn_tanh_f32(v);
#else
  // tanh(v) = 1 - 2/(e^{2v}+1); saturates correctly for |v| large (v_exp_f32 path)
  float e = __expf(2.0f * v);
  return 1.0f - 2.0f / (e + 1.0f);
#endif
}

// Fused SimpleRNN: per block = one 16-row batch tile.
// waves 0-3: 16x16 N-tiles of h_t = tanh(x_t@Wx + b + h_{t-1}@Wh) via WMMA
// wave 4   : Dense(2) projection of h_{t-1} (one step behind, double buffer)
__global__ __launch_bounds__(160) void rnn_fused(
    const float* __restrict__ x,  const float* __restrict__ Wx,
    const float* __restrict__ Wh, const float* __restrict__ bias,
    const float* __restrict__ Wd, const float* __restrict__ bd,
    float* __restrict__ out)
{
  __shared__ __align__(16) _Float16 hbuf[2][16][HP];

  const int tid  = threadIdx.x;
  const int wave = tid >> 5;
  const int lane = tid & 31;
  const int tile = blockIdx.x;

  // zero both h buffers (h_{-1} = 0)
  {
    unsigned* p = (unsigned*)&hbuf[0][0][0];
    const int ndw = (2 * 16 * HP) / 2;
    for (int i = tid; i < ndw; i += 160) p[i] = 0u;
  }
  __syncthreads();

  if (wave < 4) {
    const int  nt    = wave;
    const int  mlane = lane & 15;
    const bool hi    = lane >= 16;
    const int  col   = nt * 16 + mlane;   // global unit index of this lane's N column

    // ---- preload Wh as two f16 B operands (32x16 each, K=0..31 / 32..63) ----
    v16h whb0, whb1;
    {
      const int kb = hi ? 8 : 0;          // lanes 16-31 hold K rows 8..15 and 24..31 of each operand
      #pragma unroll
      for (int v = 0; v < 8; ++v) {
        const int kv = kb + 2*v + (v >= 4 ? 8 : 0);
        whb0[2*v]   = (_Float16)Wh[(kv    ) * UN + col];
        whb0[2*v+1] = (_Float16)Wh[(kv + 1) * UN + col];
        whb1[2*v]   = (_Float16)Wh[(32 + kv    ) * UN + col];
        whb1[2*v+1] = (_Float16)Wh[(32 + kv + 1) * UN + col];
      }
    }
    // ---- preload Wx as three f32 B operands (4x16, K chunks 0-3,4-7,8-11) ----
    // chunk 2 only has real rows k=8,9; rows 10,11 (upper lanes) are zeroed.
    v2f wxb0, wxb1, wxb2;
    {
      const int k0 = 0 + (hi ? 2 : 0);
      const int k1 = 4 + (hi ? 2 : 0);
      wxb0[0] = Wx[k0 * UN + col];       wxb0[1] = Wx[(k0 + 1) * UN + col];
      wxb1[0] = Wx[k1 * UN + col];       wxb1[1] = Wx[(k1 + 1) * UN + col];
      if (hi) { wxb2[0] = 0.0f; wxb2[1] = 0.0f; }
      else    { wxb2[0] = Wx[8 * UN + col]; wxb2[1] = Wx[9 * UN + col]; }
    }
    const float bc = bias[col];

    // ---- x pipeline: f32 A operands (16x4), lane = row M, VGPRs = K pair ----
    const float* xrow = x + (size_t)(tile * 16 + mlane) * Tn * DIN;
    const int xko = hi ? 2 : 0;
    v2f xa0 = *(const v2f*)(xrow + 0 + xko);
    v2f xa1 = *(const v2f*)(xrow + 4 + xko);
    v2f xa2 = *(const v2f*)(xrow + 8);          // k=8,9 for all lanes (B pad rows are 0)

    for (int t = 0; t < Tn; ++t) {
      if (t + 8 < Tn) __builtin_prefetch(xrow + (size_t)(t + 8) * DIN, 0, 0);
      v2f nx0 = xa0, nx1 = xa1, nx2 = xa2;
      if (t + 1 < Tn) {
        const float* xp = xrow + (size_t)(t + 1) * DIN;
        nx0 = *(const v2f*)(xp + 0 + xko);
        nx1 = *(const v2f*)(xp + 4 + xko);
        nx2 = *(const v2f*)(xp + 8);
      }

      // h_{t-1} f16 A operands straight from row-major LDS (no transpose needed)
      const _Float16* hr = &hbuf[t & 1][mlane][0];
      const int rsel = hi ? 8 : 0;
      AHalf16 a0, a1;
      a0.q[0] = *(const uint4*)(hr + rsel);        // K 0-7   (or 8-15)
      a0.q[1] = *(const uint4*)(hr + 16 + rsel);   // K 16-23 (or 24-31)
      a1.q[0] = *(const uint4*)(hr + 32 + rsel);   // K 32-39 (or 40-47)
      a1.q[1] = *(const uint4*)(hr + 48 + rsel);   // K 48-55 (or 56-63)

      v8f c;
      #pragma unroll
      for (int r = 0; r < 8; ++r) c[r] = bc;
      // input projection (independent of h -> off the recurrent critical path)
      c = __builtin_amdgcn_wmma_f32_16x16x4_f32(false, xa0, false, wxb0, (short)0, c, false, false);
      c = __builtin_amdgcn_wmma_f32_16x16x4_f32(false, xa1, false, wxb1, (short)0, c, false, false);
      c = __builtin_amdgcn_wmma_f32_16x16x4_f32(false, xa2, false, wxb2, (short)0, c, false, false);
      // recurrent matmul: the only h-dependent work per step
      c = __builtin_amdgcn_wmma_f32_16x16x32_f16(false, a0.v, false, whb0, (short)0, c, false, false);
      c = __builtin_amdgcn_wmma_f32_16x16x32_f16(false, a1.v, false, whb1, (short)0, c, false, false);

      _Float16* hw = &hbuf[1 - (t & 1)][0][0];
      const int rowoff = hi ? 8 : 0;               // C layout: lanes 16-31 hold rows M+8
      #pragma unroll
      for (int r = 0; r < 8; ++r) {
        float hv = fast_tanh(c[r]);
        hw[(size_t)(r + rowoff) * HP + col] = (_Float16)hv;
      }
      xa0 = nx0; xa1 = nx1; xa2 = nx2;
      __syncthreads();
    }
  } else {
    // ---- output wave: Dense(2) on h_{t-1}, pipelined one step behind ----
    const int row = lane & 15;
    const int o   = lane >> 4;
    float wd[UN];
    #pragma unroll
    for (int u = 0; u < UN; ++u) wd[u] = Wd[u * DOUT + o];
    const float bo = bd[o];
    float* orow = out + (size_t)(tile * 16 + row) * Tn * DOUT + o;

    for (int t = 0; t < Tn; ++t) {
      if (t > 0) {
        const _Float16* hr = &hbuf[t & 1][row][0];
        float acc = bo;
        #pragma unroll
        for (int cch = 0; cch < 8; ++cch) {
          H8 hh; hh.q = *(const uint4*)(hr + cch * 8);
          #pragma unroll
          for (int j = 0; j < 8; ++j) acc += (float)hh.h[j] * wd[cch * 8 + j];
        }
        orow[(size_t)(t - 1) * DOUT] = acc;
      }
      __syncthreads();
    }
    // final step: h_{T-1} lives in buf[0] (T even), already synced by last barrier
    {
      const _Float16* hr = &hbuf[0][row][0];
      float acc = bo;
      #pragma unroll
      for (int cch = 0; cch < 8; ++cch) {
        H8 hh; hh.q = *(const uint4*)(hr + cch * 8);
        #pragma unroll
        for (int j = 0; j < 8; ++j) acc += (float)hh.h[j] * wd[cch * 8 + j];
      }
      orow[(size_t)(Tn - 1) * DOUT] = acc;
    }
  }
}

extern "C" void kernel_launch(void* const* d_in, const int* in_sizes, int n_in,
                              void* d_out, int out_size, void* d_ws, size_t ws_size,
                              hipStream_t stream) {
  (void)in_sizes; (void)n_in; (void)out_size; (void)d_ws; (void)ws_size;
  const float* x  = (const float*)d_in[0];
  const float* Wx = (const float*)d_in[1];
  const float* Wh = (const float*)d_in[2];
  const float* b  = (const float*)d_in[3];
  const float* Wd = (const float*)d_in[4];
  const float* bd = (const float*)d_in[5];
  float* out = (float*)d_out;

  dim3 grid(Bsz / 16);   // 16 batch tiles
  dim3 block(160);       // 4 WMMA waves + 1 output wave (wave32)
  rnn_fused<<<grid, block, 0, stream>>>(x, Wx, Wh, b, Wd, bd, out);
}
